// LSTMCharTagger_62818191671891
// MI455X (gfx1250) — compile-verified, matrix operand order checked by reading, not run on hardware
//
#include <hip/hip_runtime.h>
#include <hip/hip_bf16.h>
#include <stdint.h>

// ---------------------------------------------------------------------------
// CDNA5 (gfx1250) LSTM char tagger.
// WMMA f32_16x16x32_f16 for all GEMMs AND for the recurrent h@Whh^T GEMVs
// (Whh held persistently in VGPRs as pre-packed fragments).
// ---------------------------------------------------------------------------

typedef __attribute__((ext_vector_type(16))) _Float16 v16h;
typedef __attribute__((ext_vector_type(8)))  float    v8f;
typedef __attribute__((ext_vector_type(4)))  uint32_t v4u;

union Frag { v4u u[2]; v16h h; };

#define NCH_  16384
#define NW_   2048
#define EC_   64
#define HC_   128
#define EW_   256
#define HW_   512
#define TTAG_ 64
#define KW_   384   /* EW+HC */
#define NGC_  512   /* 4*HC  */
#define NGW_  2048  /* 4*HW  */

__device__ __forceinline__ v8f wmma_f16(v16h a, v16h b, v8f c) {
  return __builtin_amdgcn_wmma_f32_16x16x32_f16(false, a, false, b, (short)0, c,
                                                false, false);
}

// Fragment-packed index for A (MxK, 16x32 tiles) per 16-bit A layout (ISA 7.12.2)
__device__ __forceinline__ size_t packA_idx(int m, int k, int K) {
  int tm = m >> 4, tk = k >> 5, mi = m & 15, ki = k & 31;
  int group = ki >> 4, r = ki & 15, hf = r >> 3, s = r & 7;
  int lane = hf * 16 + mi;
  int e = (group * 4 + (s >> 1)) * 2 + (s & 1);
  return ((size_t)(tm * (K >> 5) + tk) * 32 + lane) * 16 + e;
}
// Fragment-packed index for B (KxN, 32x16 tiles); mirrors A with n<->m roles.
__device__ __forceinline__ size_t packB_idx(int k, int n, int N) {
  int tk = k >> 5, tn = n >> 4, ni = n & 15, ki = k & 31;
  int group = ki >> 4, r = ki & 15, hf = r >> 3, s = r & 7;
  int lane = hf * 16 + ni;
  int e = (group * 4 + (s >> 1)) * 2 + (s & 1);
  return ((size_t)(tk * (N >> 4) + tn) * 32 + lane) * 16 + e;
}

__device__ __forceinline__ float sigf(float x) { return 1.f / (1.f + __expf(-x)); }

// ---------------------------------------------------------------------------
// Pack kernels
// ---------------------------------------------------------------------------
// B from row-major weight W [N x K]:  B[k][n] = W[n][k]
__global__ void pack_b_f32(const float* __restrict__ W, _Float16* __restrict__ dst,
                           int N, int K) {
  int idx = blockIdx.x * 256 + threadIdx.x;
  if (idx >= N * K) return;
  int n = idx / K, k = idx % K;
  dst[packB_idx(k, n, N)] = (_Float16)W[(size_t)n * K + k];
}
// A from row-major f32 src [M x K]
__global__ void pack_a_f32(const float* __restrict__ src, _Float16* __restrict__ dst,
                           int M, int K) {
  int idx = blockIdx.x * 256 + threadIdx.x;
  if (idx >= M * K) return;
  int m = idx / K, k = idx % K;
  dst[packA_idx(m, k, K)] = (_Float16)src[(size_t)m * K + k];
}
// char_x[i,k] = char_emb[char_ids[i], k]   (M=NCH, K=EC)
__global__ void pack_charx(const int* __restrict__ ids, const float* __restrict__ emb,
                           _Float16* __restrict__ dst) {
  int idx = blockIdx.x * 256 + threadIdx.x;
  int m = idx >> 6, k = idx & 63;
  dst[packA_idx(m, k, EC_)] = (_Float16)emb[(size_t)ids[m] * EC_ + k];
}
// wx[i,k] = k<EW ? word_emb[word_ids[i],k] : hs_char[last_idx[i], k-EW]
__global__ void pack_wx(const int* __restrict__ wids, const int* __restrict__ lidx,
                        const float* __restrict__ wemb, const float* __restrict__ hs_c,
                        _Float16* __restrict__ dst) {
  int idx = blockIdx.x * 256 + threadIdx.x;
  int m = idx / KW_, k = idx % KW_;
  float v = (k < EW_) ? wemb[(size_t)wids[m] * EW_ + k]
                      : hs_c[(size_t)lidx[m] * HC_ + (k - EW_)];
  dst[packA_idx(m, k, KW_)] = (_Float16)v;
}

// ---------------------------------------------------------------------------
// Fragment GEMM: C[MxN] = A(pack) x B(pack) + bias, 1 wave = one 16x16 tile
// ---------------------------------------------------------------------------
__global__ __launch_bounds__(256) void wmma_gemm(
    const _Float16* __restrict__ A, const _Float16* __restrict__ B,
    const float* __restrict__ bias, float* __restrict__ C,
    int Mt, int Nt, int Kt) {
  int wave = threadIdx.x >> 5, lane = threadIdx.x & 31;
  int tile = blockIdx.x * 8 + wave;
  if (tile >= Mt * Nt) return;          // uniform per-wave: EXEC stays all-ones
  int tm = tile / Nt, tn = tile % Nt;
  v8f acc = {};
  for (int kt = 0; kt < Kt; ++kt) {
    Frag a, b;
    const v4u* pa = (const v4u*)A + (((size_t)(tm * Kt + kt) * 32 + lane) << 1);
    const v4u* pb = (const v4u*)B + (((size_t)(kt * Nt + tn) * 32 + lane) << 1);
    a.u[0] = pa[0]; a.u[1] = pa[1];
    b.u[0] = pb[0]; b.u[1] = pb[1];
    acc = wmma_f16(a.h, b.h, acc);
  }
  int N = Nt * 16;
  int n = tn * 16 + (lane & 15);
  int mb = tm * 16 + ((lane >> 4) << 3);
  float bv = bias[n];
#pragma unroll
  for (int r = 0; r < 8; ++r) C[(size_t)(mb + r) * N + n] = acc[r] + bv;
}

// ---------------------------------------------------------------------------
// Char LSTM scan: 1 workgroup (16 waves). Whh_c^T fragments live in VGPRs.
// Per step: h (replicated across A rows) x Whh -> gates, then pointwise LSTM.
// ---------------------------------------------------------------------------
__global__ __launch_bounds__(512, 1) void char_scan(
    const _Float16* __restrict__ Bp,   // packed Whh_c^T  (Kt=4, Nt=32)
    const float* __restrict__ xg,      // [NCH x 512] (bias already included)
    float* __restrict__ hs) {          // [NCH x 128]
  __shared__ __align__(16) uint32_t h_lds[HC_ / 2];  // 128 f16
  __shared__ float gbuf[NGC_];
  __shared__ float xbuf[NGC_];
  int tid = threadIdx.x, lane = tid & 31, wave = tid >> 5;
  int lhalf = lane >> 4;

  Frag bf[8];                          // 2 N-tiles x 4 K-tiles per wave
#pragma unroll
  for (int kt = 0; kt < 4; ++kt)
#pragma unroll
    for (int j = 0; j < 2; ++j) {
      int tn = 2 * wave + j;
      const v4u* p = (const v4u*)Bp + (((size_t)(kt * 32 + tn) * 32 + lane) << 1);
      bf[kt * 2 + j].u[0] = p[0];
      bf[kt * 2 + j].u[1] = p[1];
    }
  if (tid < HC_ / 2) h_lds[tid] = 0u;
  float c_state = 0.f;
  float xgv = xg[tid];                 // prefetch t=0
  __syncthreads();

  for (int t = 0; t < NCH_; ++t) {
    xbuf[tid] = xgv;
    v8f accA = {}, accB = {};
    const v4u* hp = (const v4u*)h_lds;
#pragma unroll
    for (int kt = 0; kt < 4; ++kt) {
      Frag a;
      a.u[0] = hp[(kt * 16 + lhalf * 4) >> 2];
      a.u[1] = hp[(kt * 16 + 8 + lhalf * 4) >> 2];
      accA = wmma_f16(a.h, bf[kt * 2].h, accA);
      accB = wmma_f16(a.h, bf[kt * 2 + 1].h, accB);
    }
    gbuf[wave * 32 + lane] = (lane < 16) ? accA[0] : accB[0];
    if (t + 1 < NCH_) xgv = xg[(size_t)(t + 1) * NGC_ + tid];  // prefetch
    __syncthreads();
    if (tid < HC_) {
      float gi = gbuf[tid] + xbuf[tid];
      float gf = gbuf[HC_ + tid] + xbuf[HC_ + tid];
      float gg = gbuf[2 * HC_ + tid] + xbuf[2 * HC_ + tid];
      float go = gbuf[3 * HC_ + tid] + xbuf[3 * HC_ + tid];
      float fi = sigf(gi), ff = sigf(gf), fo = sigf(go);
      c_state = ff * c_state + fi * tanhf(gg);
      float h = fo * tanhf(c_state);
      hs[(size_t)t * HC_ + tid] = h;
      ((_Float16*)h_lds)[tid] = (_Float16)h;
    }
    __syncthreads();
  }
}

// ---------------------------------------------------------------------------
// Global sense-reversing barrier for the 16-WG word scan
// ---------------------------------------------------------------------------
__device__ __forceinline__ void gbar(int* bar, int nwg, int* lsense, int tid) {
  __syncthreads();
  if (tid == 0) {
    int prev = __hip_atomic_fetch_add(&bar[0], 1, __ATOMIC_ACQ_REL,
                                      __HIP_MEMORY_SCOPE_AGENT);
    if (prev == nwg - 1) {
      __hip_atomic_store(&bar[0], 0, __ATOMIC_RELAXED, __HIP_MEMORY_SCOPE_AGENT);
      __hip_atomic_fetch_add(&bar[16], 1, __ATOMIC_RELEASE,
                             __HIP_MEMORY_SCOPE_AGENT);
    }
    (*lsense)++;
    while (__hip_atomic_load(&bar[16], __ATOMIC_ACQUIRE,
                             __HIP_MEMORY_SCOPE_AGENT) < *lsense)
      __builtin_amdgcn_s_sleep(2);
  }
  __syncthreads();
}

// ---------------------------------------------------------------------------
// Word LSTM scan: 16 WGs x 512 threads. WG p owns gate rows b*512+p*32..+31
// for all 4 gate blocks b (8 N-tiles); Whh_w^T slice resident in VGPRs
// (split-K across wave pairs). h is double-buffered f16 through L2.
// ---------------------------------------------------------------------------
__global__ __launch_bounds__(512, 1) void word_scan(
    const _Float16* __restrict__ Bp,   // packed Whh_w^T  (Kt=16, Nt=128)
    const float* __restrict__ xg,      // [NW x 2048] (bias included)
    float* __restrict__ hs,            // [NW x 512]
    _Float16* __restrict__ hbuf,       // [2][512]
    int* __restrict__ bar) {
  __shared__ __align__(16) uint32_t h_lds[HW_ / 2];   // 512 f16
  __shared__ float pbuf[256];
  __shared__ float gsum[128];
  int tid = threadIdx.x, lane = tid & 31, wave = tid >> 5, p = blockIdx.x;
  int lhalf = lane >> 4;
  int j = wave & 7, khalf = wave >> 3;
  int gt = (j >> 1) * 32 + 2 * p + (j & 1);           // global N-tile

  Frag bf[8];                                         // 8 K-tiles of one N-tile
#pragma unroll
  for (int q = 0; q < 8; ++q) {
    int kt = khalf * 8 + q;
    const v4u* ptr = (const v4u*)Bp + (((size_t)(kt * 128 + gt) * 32 + lane) << 1);
    bf[q].u[0] = ptr[0];
    bf[q].u[1] = ptr[1];
  }
  if (tid < 32) hbuf[p * 32 + tid] = (_Float16)0.f;   // zero h buffer 0 slice
  float c_state = 0.f;
  int grow = 0;
  float xgv = 0.f;
  if (tid < 128) {
    int jj = tid >> 4, nn = tid & 15;
    grow = (jj >> 1) * HW_ + p * 32 + (jj & 1) * 16 + nn;
    xgv = xg[grow];                                   // prefetch t=0
  }
  __threadfence();
  int lsense = 0;
  gbar(bar, 16, &lsense, tid);

  for (int t = 0; t < NW_; ++t) {
    const uint32_t* hc = (const uint32_t*)(hbuf + (t & 1) * HW_);
    if (tid < HW_ / 2) h_lds[tid] = hc[tid];
    __syncthreads();
    v8f acc = {};
    const v4u* hp = (const v4u*)h_lds;
#pragma unroll
    for (int q = 0; q < 8; ++q) {
      int kt = khalf * 8 + q;
      Frag a;
      a.u[0] = hp[(kt * 16 + lhalf * 4) >> 2];
      a.u[1] = hp[(kt * 16 + 8 + lhalf * 4) >> 2];
      acc = wmma_f16(a.h, bf[q].h, acc);
    }
    if (lane < 16) pbuf[wave * 16 + lane] = acc[0];
    __syncthreads();
    if (tid < 128) gsum[tid] = pbuf[tid] + pbuf[128 + tid] + xgv;
    __syncthreads();
    if (tid < 32) {
      float gi = gsum[tid], gf = gsum[32 + tid];
      float gg = gsum[64 + tid], go = gsum[96 + tid];
      float fi = sigf(gi), ff = sigf(gf), fo = sigf(go);
      c_state = ff * c_state + fi * tanhf(gg);
      float h = fo * tanhf(c_state);
      hs[(size_t)t * HW_ + p * 32 + tid] = h;
      hbuf[((t + 1) & 1) * HW_ + p * 32 + tid] = (_Float16)h;
    }
    if (tid < 128 && t + 1 < NW_) xgv = xg[(size_t)(t + 1) * NGW_ + grow];
    __threadfence();
    gbar(bar, 16, &lsense, tid);
  }
}

// ---------------------------------------------------------------------------
__global__ void init_bar(int* bar) { if (threadIdx.x < 32) bar[threadIdx.x] = 0; }

__global__ __launch_bounds__(256) void logsoftmax64(const float* __restrict__ lg,
                                                    float* __restrict__ out) {
  int row = blockIdx.x * 8 + (threadIdx.x >> 5);
  int lane = threadIdx.x & 31;
  float x0 = lg[(size_t)row * 64 + lane];
  float x1 = lg[(size_t)row * 64 + 32 + lane];
  float m = fmaxf(x0, x1);
#pragma unroll
  for (int o = 16; o > 0; o >>= 1) m = fmaxf(m, __shfl_xor(m, o, 32));
  float s = __expf(x0 - m) + __expf(x1 - m);
#pragma unroll
  for (int o = 16; o > 0; o >>= 1) s += __shfl_xor(s, o, 32);
  float ls = __logf(s);
  out[(size_t)row * 64 + lane] = x0 - m - ls;
  out[(size_t)row * 64 + 32 + lane] = x1 - m - ls;
}

// ---------------------------------------------------------------------------
extern "C" void kernel_launch(void* const* d_in, const int* in_sizes, int n_in,
                              void* d_out, int out_size, void* d_ws, size_t ws_size,
                              hipStream_t stream) {
  const int*   char_ids = (const int*)d_in[0];
  const int*   word_ids = (const int*)d_in[1];
  const int*   last_idx = (const int*)d_in[2];
  const float* word_emb = (const float*)d_in[3];
  const float* char_emb = (const float*)d_in[4];
  const float* Wih_c    = (const float*)d_in[5];
  const float* Whh_c    = (const float*)d_in[6];
  const float* b_c      = (const float*)d_in[7];
  const float* Wih_w    = (const float*)d_in[8];
  const float* Whh_w    = (const float*)d_in[9];
  const float* b_w      = (const float*)d_in[10];
  const float* W_tag    = (const float*)d_in[11];
  const float* b_tag    = (const float*)d_in[12];
  float* out = (float*)d_out;

  char* ws = (char*)d_ws;
  size_t o = 0;
  auto al = [&](size_t b) { size_t r = o; o = (o + b + 255) & ~(size_t)255; return r; };
  size_t o_pB_ihc = al((size_t)EC_ * NGC_ * 2);
  size_t o_pB_hhc = al((size_t)HC_ * NGC_ * 2);
  size_t o_pB_ihw = al((size_t)KW_ * NGW_ * 2);
  size_t o_pB_hhw = al((size_t)HW_ * NGW_ * 2);
  size_t o_pB_tag = al((size_t)HW_ * TTAG_ * 2);
  size_t o_pA_cx  = al((size_t)NCH_ * EC_ * 2);
  size_t o_xg_c   = al((size_t)NCH_ * NGC_ * 4);
  size_t o_hs_c   = al((size_t)NCH_ * HC_ * 4);
  size_t o_pA_wx  = al((size_t)NW_ * KW_ * 2);
  size_t o_xg_w   = al((size_t)NW_ * NGW_ * 4);
  size_t o_hs_w   = al((size_t)NW_ * HW_ * 4);
  size_t o_pA_hw  = al((size_t)NW_ * HW_ * 2);
  size_t o_logits = al((size_t)NW_ * TTAG_ * 4);
  size_t o_hbuf   = al((size_t)2 * HW_ * 2);
  size_t o_bar    = al(64 * 4);
  if (ws_size < o) return;  // not enough scratch

  _Float16* pB_ihc = (_Float16*)(ws + o_pB_ihc);
  _Float16* pB_hhc = (_Float16*)(ws + o_pB_hhc);
  _Float16* pB_ihw = (_Float16*)(ws + o_pB_ihw);
  _Float16* pB_hhw = (_Float16*)(ws + o_pB_hhw);
  _Float16* pB_tag = (_Float16*)(ws + o_pB_tag);
  _Float16* pA_cx  = (_Float16*)(ws + o_pA_cx);
  float*    xg_c   = (float*)(ws + o_xg_c);
  float*    hs_c   = (float*)(ws + o_hs_c);
  _Float16* pA_wx  = (_Float16*)(ws + o_pA_wx);
  float*    xg_w   = (float*)(ws + o_xg_w);
  float*    hs_w   = (float*)(ws + o_hs_w);
  _Float16* pA_hw  = (_Float16*)(ws + o_pA_hw);
  float*    logits = (float*)(ws + o_logits);
  _Float16* hbuf   = (_Float16*)(ws + o_hbuf);
  int*      bar    = (int*)(ws + o_bar);

  init_bar<<<1, 32, 0, stream>>>(bar);

  // ---- weight packing (one-off) ----
  pack_b_f32<<<(NGC_ * EC_) / 256, 256, 0, stream>>>(Wih_c, pB_ihc, NGC_, EC_);
  pack_b_f32<<<(NGC_ * HC_) / 256, 256, 0, stream>>>(Whh_c, pB_hhc, NGC_, HC_);
  pack_b_f32<<<(NGW_ * KW_) / 256, 256, 0, stream>>>(Wih_w, pB_ihw, NGW_, KW_);
  pack_b_f32<<<(NGW_ * HW_) / 256, 256, 0, stream>>>(Whh_w, pB_hhw, NGW_, HW_);
  pack_b_f32<<<(TTAG_ * HW_) / 256, 256, 0, stream>>>(W_tag, pB_tag, TTAG_, HW_);

  // ---- char pipeline ----
  pack_charx<<<(NCH_ * EC_) / 256, 256, 0, stream>>>(char_ids, char_emb, pA_cx);
  wmma_gemm<<<(NCH_ / 16) * (NGC_ / 16) / 8, 256, 0, stream>>>(
      pA_cx, pB_ihc, b_c, xg_c, NCH_ / 16, NGC_ / 16, EC_ / 32);
  char_scan<<<1, 512, 0, stream>>>(pB_hhc, xg_c, hs_c);

  // ---- word pipeline ----
  pack_wx<<<(NW_ * KW_) / 256, 256, 0, stream>>>(word_ids, last_idx, word_emb,
                                                 hs_c, pA_wx);
  wmma_gemm<<<(NW_ / 16) * (NGW_ / 16) / 8, 256, 0, stream>>>(
      pA_wx, pB_ihw, b_w, xg_w, NW_ / 16, NGW_ / 16, KW_ / 32);
  word_scan<<<16, 512, 0, stream>>>(pB_hhw, xg_w, hs_w, hbuf, bar);

  // ---- tagger head ----
  pack_a_f32<<<(NW_ * HW_) / 256, 256, 0, stream>>>(hs_w, pA_hw, NW_, HW_);
  wmma_gemm<<<(NW_ / 16) * (TTAG_ / 16) / 8, 256, 0, stream>>>(
      pA_hw, pB_tag, b_tag, logits, NW_ / 16, TTAG_ / 16, HW_ / 32);
  logsoftmax64<<<NW_ / 8, 256, 0, stream>>>(logits, out);
}
